// MemoryGraph_20109036880203
// MI455X (gfx1250) — matrix operand order repaired and checked by reading, not compiled, not run on hardware
//
#include <hip/hip_runtime.h>

typedef _Float16 half_t;
typedef __attribute__((ext_vector_type(16))) half_t v16h;
typedef __attribute__((ext_vector_type(8)))  float  v8f;

#define BSZ     8
#define NCELL   256
#define NCN     128    // Nc (neurons per cell)
#define DN      64     // D_n
#define HSD     256    // HS
#define HMD     128    // HM
#define ALPHA_N 2
#define GMAX    0.5f
#define EPSV    1e-6f

// flat output offsets (float32 elements), return order: h_new, m, W_new, decay_new, hebbian_new
#define OFF_M     ((size_t)16777216)
#define OFF_W     ((size_t)33554432)
#define OFF_DECAY ((size_t)67108864)
#define OFF_HEB   ((size_t)67371008)

// ---- LDS layout (bytes), total 304 KB dynamic ----
// [0      ,16K ) msg_w1 f16 [HM][DN]        -> reused as m16 f16 [Nc][DN] after GEMM1
// [16K    ,32K ) msg_w2 f16 [DN][HM]
// [32K    ,96K ) state_w1 f16 [HS][128]
// [96K    ,128K) state_w2 f16 [DN][HS]
// [128K   ,144K) hf16 f16 [Nc][DN]           (h after injection)
// [144K   ,176K) cat16 f16 [Nc][128]  (h_id | agg)   \ 64KB -> async hebbian f32 / Wtmp f32 [Nc][Nc]
// [176K   ,208K) W16 f16 [Nc][Nc]                    /
// [208K   ,240K) th1 f16 [Nc][HM]            -> reused: injbuf f32[128] (phase0), dh f32 [Nc][DN]
// [240K   ,304K) th2 f16 [Nc][HS]            -> reused: hnew16 f16 [Nc][DN] @240K, hg/wg rows @256K
#define LDS_BYTES   (304 * 1024)
#define HEB_LDS_OFF (144 * 1024)

__device__ __forceinline__ float sigmoid_fast(float x) {
  return __builtin_amdgcn_rcpf(1.0f + __expf(-x));
}

// branch-free tanh: (e-1)/(e+1), e = exp(2*clamp(x)); exact enough for post-GEMM activations
__device__ __forceinline__ float tanh_fast(float x) {
  const float cx = fminf(fmaxf(x, -9.0f), 9.0f);
  const float e  = __expf(2.0f * cx);
  return (e - 1.0f) * __builtin_amdgcn_rcpf(e + 1.0f);
}

// A fragment: 16x32 f16, M rows striped on lanes 0-15 (=lanes 16-31), K contiguous: k = 16*half + s
__device__ __forceinline__ v16h load_A_frag(const half_t* base, int ld, int m0, int k0, int lane) {
  const int m = m0 + (lane & 15);
  const int g = lane >> 4;
  const half_t* row = base + m * ld + k0;
  v16h a;
#pragma unroll
  for (int t = 0; t < 8; ++t) {
    const int k = 2 * t + 16 * g;
    a[2 * t]     = row[k];
    a[2 * t + 1] = row[k + 1];
  }
  return a;
}

// B fragment from a [N][K] row-major matrix (i.e. B = Wt^T): B[k][n] = Wt[n][k]
__device__ __forceinline__ v16h load_BT_frag(const half_t* base, int ld, int n0, int k0, int lane) {
  const int n = n0 + (lane & 15);
  const int g = lane >> 4;
  const half_t* row = base + n * ld + k0 + 16 * g;
  v16h b;
#pragma unroll
  for (int i = 0; i < 16; ++i) b[i] = row[i];
  return b;
}

// B fragment from a [K][N] row-major matrix
__device__ __forceinline__ v16h load_B_frag(const half_t* base, int ld, int k0, int n0, int lane) {
  const int n = n0 + (lane & 15);
  const int g = lane >> 4;
  const half_t* col = base + (k0 + 16 * g) * ld + n;
  v16h b;
#pragma unroll
  for (int i = 0; i < 16; ++i) b[i] = col[i * ld];
  return b;
}

__device__ __forceinline__ v8f wmma16(v16h a, v16h b, v8f c) {
  return __builtin_amdgcn_wmma_f32_16x16x32_f16(false, a, false, b, (short)0, c, false, false);
}

__global__ void __launch_bounds__(256)
memgraph_cell_kernel(const float* __restrict__ x, const float* __restrict__ h,
                     const float* __restrict__ W, const float* __restrict__ decay,
                     const float* __restrict__ hebbian, const float* __restrict__ neuron_id,
                     const float* __restrict__ state_w1, const float* __restrict__ state_b1,
                     const float* __restrict__ state_w2, const float* __restrict__ state_b2,
                     const float* __restrict__ msg_w1, const float* __restrict__ msg_b1,
                     const float* __restrict__ msg_w2, const float* __restrict__ msg_b2,
                     const float* __restrict__ inject_w, const float* __restrict__ inject_b,
                     const float* __restrict__ W_decay_logit, const float* __restrict__ decay_gamma_logit,
                     const float* __restrict__ hebbian_decay_logit,
                     float* __restrict__ out) {
  extern __shared__ char lds[];
  half_t* msgw1  = (half_t*)(lds + 0);
  half_t* m16    = msgw1;                       // alias, live after GEMM2
  half_t* msgw2  = (half_t*)(lds + 16 * 1024);
  half_t* stw1   = (half_t*)(lds + 32 * 1024);
  half_t* stw2   = (half_t*)(lds + 96 * 1024);
  half_t* hf16   = (half_t*)(lds + 128 * 1024);
  half_t* cat16  = (half_t*)(lds + 144 * 1024);
  half_t* W16    = (half_t*)(lds + 176 * 1024);
  float*  Wtmp   = (float*)(lds + HEB_LDS_OFF); // alias cat16+W16: async hebbian, then W_tmp
  half_t* th1    = (half_t*)(lds + 208 * 1024);
  float*  injbuf = (float*)(lds + 208 * 1024);  // alias, phase-0 only
  float*  dhbuf  = (float*)(lds + 208 * 1024);  // alias, live after GEMM5
  half_t* th2    = (half_t*)(lds + 240 * 1024);
  half_t* hnew16 = (half_t*)(lds + 240 * 1024); // alias, live after elementwise
  float*  hg_row = (float*)(lds + 256 * 1024);
  float*  wg_row = hg_row + NCN;

  const int bc   = blockIdx.x;            // b*NC + c
  const int c    = bc & (NCELL - 1);
  const int tid  = threadIdx.x;
  const int lane = tid & 31;
  const int wave = tid >> 5;
  const int g    = lane >> 4;
  const int nl   = lane & 15;

  // ---- phase 0: input injection (128 matvecs of length 64) ----
  if (tid < ALPHA_N * DN) {
    const float* iw = inject_w + ((size_t)c * ALPHA_N * DN + tid) * DN;
    const float* xv = x + (size_t)bc * DN;
    float s = inject_b[c * ALPHA_N * DN + tid];
    for (int d = 0; d < DN; ++d) s += xv[d] * iw[d];
    injbuf[tid] = s;
  }
  __syncthreads();

  // ---- phase 1: stage h(+inj), h_id, W, weights into LDS (f16) ----
  const float* hG  = h + (size_t)bc * NCN * DN;
  const float* nid = neuron_id + (size_t)c * NCN * DN;
  for (int idx = tid; idx < NCN * DN; idx += 256) {
    const int n = idx >> 6, d = idx & 63;
    float hv = hG[idx];
    if (n < ALPHA_N) hv += injbuf[n * DN + d];
    hf16[idx] = (half_t)hv;
    cat16[n * 128 + d] = (half_t)(hv + nid[idx]);
  }
  const float* WG = W + (size_t)bc * NCN * NCN;
  for (int idx = tid; idx < NCN * NCN; idx += 256) W16[idx] = (half_t)WG[idx];
  for (int idx = tid; idx < HMD * DN; idx += 256) msgw1[idx] = (half_t)msg_w1[idx];
  for (int idx = tid; idx < DN * HMD; idx += 256) msgw2[idx] = (half_t)msg_w2[idx];
  for (int idx = tid; idx < HSD * 128; idx += 256) stw1[idx] = (half_t)state_w1[idx];
  for (int idx = tid; idx < DN * HSD; idx += 256) stw2[idx] = (half_t)state_w2[idx];
  __syncthreads();

  // ---- GEMM1: th1[n,hm] = tanh(h_id . msg_w1^T + b1)   M=128 N=128 K=64 ----
  for (int t = wave; t < 64; t += 8) {
    const int m0 = (t >> 3) << 4, n0 = (t & 7) << 4;
    const float bv = msg_b1[n0 + nl];
    v8f acc = {bv, bv, bv, bv, bv, bv, bv, bv};
#pragma unroll
    for (int k0 = 0; k0 < DN; k0 += 32)
      acc = wmma16(load_A_frag(cat16, 128, m0, k0, lane),
                   load_BT_frag(msgw1, DN, n0, k0, lane), acc);
    const int rb = m0 + 8 * g, col = n0 + nl;
#pragma unroll
    for (int r = 0; r < 8; ++r) th1[(rb + r) * HMD + col] = (half_t)tanh_fast(acc[r]);
  }
  __syncthreads();

  // ---- GEMM2: m[n,d] = th1 . msg_w2^T + b2   M=128 N=64 K=128 -> global + m16 ----
  const size_t mOff = OFF_M + (size_t)bc * NCN * DN;
  for (int t = wave; t < 32; t += 8) {
    const int m0 = (t >> 2) << 4, n0 = (t & 3) << 4;
    const float bv = msg_b2[n0 + nl];
    v8f acc = {bv, bv, bv, bv, bv, bv, bv, bv};
#pragma unroll
    for (int k0 = 0; k0 < HMD; k0 += 32)
      acc = wmma16(load_A_frag(th1, HMD, m0, k0, lane),
                   load_BT_frag(msgw2, HMD, n0, k0, lane), acc);
    const int rb = m0 + 8 * g, col = n0 + nl;
#pragma unroll
    for (int r = 0; r < 8; ++r) {
      const float v = acc[r];
      out[mOff + (size_t)(rb + r) * DN + col] = v;
      m16[(rb + r) * DN + col] = (half_t)v;
    }
  }
  __syncthreads();

  // ---- GEMM3: agg = W . m   M=128 N=64 K=128 -> cat16 cols 64..127 ----
  for (int t = wave; t < 32; t += 8) {
    const int m0 = (t >> 2) << 4, n0 = (t & 3) << 4;
    v8f acc = {};
#pragma unroll
    for (int k0 = 0; k0 < NCN; k0 += 32)
      acc = wmma16(load_A_frag(W16, NCN, m0, k0, lane),
                   load_B_frag(m16, DN, k0, n0, lane), acc);
    const int rb = m0 + 8 * g, col = n0 + nl;
#pragma unroll
    for (int r = 0; r < 8; ++r) cat16[(rb + r) * 128 + 64 + col] = (half_t)acc[r];
  }
  __syncthreads();

  // ---- GEMM4: th2 = tanh(cat . state_w1^T + b1)   M=128 N=256 K=128 ----
  for (int t = wave; t < 128; t += 8) {
    const int m0 = (t >> 4) << 4, n0 = (t & 15) << 4;
    const float bv = state_b1[n0 + nl];
    v8f acc = {bv, bv, bv, bv, bv, bv, bv, bv};
#pragma unroll
    for (int k0 = 0; k0 < 128; k0 += 32)
      acc = wmma16(load_A_frag(cat16, 128, m0, k0, lane),
                   load_BT_frag(stw1, 128, n0, k0, lane), acc);
    const int rb = m0 + 8 * g, col = n0 + nl;
#pragma unroll
    for (int r = 0; r < 8; ++r) th2[(rb + r) * HSD + col] = (half_t)tanh_fast(acc[r]);
  }
  __syncthreads();

  // ---- kick off async DMA of the hebbian tile (64KB) into the now-dead cat16/W16 region;
  //      overlaps GEMM5 + elementwise. ASYNCcnt tracked, waited before the GEMM6 epilogue. ----
  {
    const char* hebBytes = (const char*)(hebbian + (size_t)bc * NCN * NCN);
    for (int i = tid; i < (NCN * NCN * 4) / 16; i += 256) {   // 4096 x b128 transfers
      const unsigned      ldsOff = HEB_LDS_OFF + (unsigned)i * 16;
      const unsigned long long ga = (unsigned long long)(hebBytes + (size_t)i * 16);
      asm volatile("global_load_async_to_lds_b128 %0, %1, off"
                   :: "v"(ldsOff), "v"(ga)
                   : "memory");
    }
  }

  // ---- GEMM5: dh = th2 . state_w2^T + b2   M=128 N=64 K=256 -> dhbuf f32 ----
  for (int t = wave; t < 32; t += 8) {
    const int m0 = (t >> 2) << 4, n0 = (t & 3) << 4;
    const float bv = state_b2[n0 + nl];
    v8f acc = {bv, bv, bv, bv, bv, bv, bv, bv};
#pragma unroll
    for (int k0 = 0; k0 < HSD; k0 += 32)
      acc = wmma16(load_A_frag(th2, HSD, m0, k0, lane),
                   load_BT_frag(stw2, HSD, n0, k0, lane), acc);
    const int rb = m0 + 8 * g, col = n0 + nl;
#pragma unroll
    for (int r = 0; r < 8; ++r) dhbuf[(rb + r) * DN + col] = acc[r];
  }
  __syncthreads();

  // ---- elementwise: decay_new, h_new, gate rows ----
  const float* dec = decay + (size_t)bc * NCN;
  const float* dgl = decay_gamma_logit + (size_t)c * NCN;
  const float* hgl = hebbian_decay_logit + (size_t)c * NCN;
  const float* wgl = W_decay_logit + (size_t)c * NCN;
  if (tid < NCN) {
    const int n = tid;
    float s = 0.0f;
    for (int d = 0; d < DN; ++d) s += dhbuf[n * DN + ((d + n) & 63)]; // skewed, bank-friendly
    const float dg = GMAX * sigmoid_fast(dgl[n]);
    const float dn = (1.0f - dg) * dec[n] + dg * sigmoid_fast(s * (1.0f / DN));
    out[OFF_DECAY + (size_t)bc * NCN + n] = dn;
    const size_t hOff = (size_t)bc * NCN * DN + (size_t)n * DN; // OFF_H == 0
    for (int d = 0; d < DN; ++d) {
      const float hn = (1.0f - dn) * (float)hf16[n * DN + d] + dn * tanh_fast(dhbuf[n * DN + d]);
      out[hOff + d] = hn;
      hnew16[n * DN + d] = (half_t)hn;
    }
    hg_row[n] = GMAX * sigmoid_fast(hgl[n]);
    wg_row[n] = GMAX * sigmoid_fast(wgl[n]);
  }
  // each wave drains its own async loads, then barrier makes all hebbian data visible
  asm volatile("s_wait_asynccnt 0x0" ::: "memory");
  __syncthreads();

  // ---- GEMM6: outer = h_new . m^T / 64; fuse hebbian_new (LDS) + pre-norm W (in-place) ----
  const size_t hebOff = OFF_HEB + (size_t)bc * NCN * NCN;
  for (int t = wave; t < 64; t += 8) {
    const int m0 = (t >> 3) << 4, n0 = (t & 7) << 4;
    v8f acc = {};
#pragma unroll
    for (int k0 = 0; k0 < DN; k0 += 32)
      acc = wmma16(load_A_frag(hnew16, DN, m0, k0, lane),
                   load_BT_frag(m16, DN, n0, k0, lane), acc);
    const int rb = m0 + 8 * g, col = n0 + nl;
#pragma unroll
    for (int r = 0; r < 8; ++r) {
      const int row = rb + r;
      const int ei  = row * NCN + col;
      const float hgv = hg_row[row];
      const float hn  = (1.0f - hgv) * Wtmp[ei] + hgv * (acc[r] * (1.0f / DN)); // Wtmp holds hebbian (async)
      out[hebOff + (size_t)ei] = hn;
      const float wgv = wg_row[row];
      Wtmp[ei] = (1.0f - wgv) * WG[ei] + wgv * hn;   // in-place: heb value already consumed
    }
  }
  __syncthreads();

  // ---- rms_norm over incoming-edge axis, write W_new ----
  const size_t wOff = OFF_W + (size_t)bc * NCN * NCN;
  if (tid < NCN) {
    const int i = tid;
    float ss = 0.0f;
    for (int j = 0; j < NCN; ++j) {
      const float v = Wtmp[i * NCN + ((j + i) & 127)]; // skewed read
      ss += v * v;
    }
    const float rs = rsqrtf(ss * (1.0f / NCN) + EPSV);
    for (int j = 0; j < NCN; ++j) out[wOff + (size_t)i * NCN + j] = Wtmp[i * NCN + j] * rs;
  }
}

extern "C" void kernel_launch(void* const* d_in, const int* in_sizes, int n_in,
                              void* d_out, int out_size, void* d_ws, size_t ws_size,
                              hipStream_t stream) {
  (void)in_sizes; (void)n_in; (void)out_size; (void)d_ws; (void)ws_size;
  const float* x    = (const float*)d_in[0];
  const float* h    = (const float*)d_in[1];
  const float* W    = (const float*)d_in[2];
  const float* dec  = (const float*)d_in[3];
  const float* heb  = (const float*)d_in[4];
  const float* nid  = (const float*)d_in[5];
  const float* sw1  = (const float*)d_in[6];
  const float* sb1  = (const float*)d_in[7];
  const float* sw2  = (const float*)d_in[8];
  const float* sb2  = (const float*)d_in[9];
  const float* mw1  = (const float*)d_in[10];
  const float* mb1  = (const float*)d_in[11];
  const float* mw2  = (const float*)d_in[12];
  const float* mb2  = (const float*)d_in[13];
  const float* iw   = (const float*)d_in[14];
  const float* ib   = (const float*)d_in[15];
  const float* wdl  = (const float*)d_in[16];
  const float* dgl  = (const float*)d_in[17];
  const float* hdl  = (const float*)d_in[18];
  float* out = (float*)d_out;

  static bool attr_set = false;
  if (!attr_set) {
    hipFuncSetAttribute((const void*)memgraph_cell_kernel,
                        hipFuncAttributeMaxDynamicSharedMemorySize, LDS_BYTES);
    attr_set = true;
  }
  dim3 grid(BSZ * NCELL);
  dim3 block(256);
  memgraph_cell_kernel<<<grid, block, LDS_BYTES, stream>>>(
      x, h, W, dec, heb, nid, sw1, sb1, sw2, sb2,
      mw1, mb1, mw2, mb2, iw, ib, wdl, dgl, hdl, out);
}
